// CPDecomposedWeights_2903397893022
// MI455X (gfx1250) — matrix operand order repaired and checked by reading, not compiled, not run on hardware
//
#include <hip/hip_runtime.h>
#include <hip/hip_bf16.h>

typedef _Float16 v16h __attribute__((ext_vector_type(16)));
typedef _Float16 v8h  __attribute__((ext_vector_type(8)));
typedef float    v8f  __attribute__((ext_vector_type(8)));
typedef float    v2f  __attribute__((ext_vector_type(2)));

#define LDS_PITCH 72   // 64 K-halfs + 8 pad halfs -> 144B row pitch, conflict-spread

__device__ __forceinline__ v8h ld8(const _Float16* p) {
    return *reinterpret_cast<const v8h*>(p);
}
__device__ __forceinline__ v16h cat(v8h lo, v8h hi) {
    v16h r;
#pragma unroll
    for (int i = 0; i < 8; ++i) { r[i] = lo[i]; r[i + 8] = hi[i]; }
    return r;
}

// out[a,b,c,d] complex64, a in [0,64), b in [0,64), c in [0,128), d in [0,65)
// row = a*64+b (4096 rows), col = c*65+d (8320 cols)
// out = sum_e (core*fa ⊙ fb)[row,e] * (fc ⊙ fd)[col,e]   (complex)
__global__ __launch_bounds__(256)
void cp_reconstruct_wmma(const float* __restrict__ core,
                         const float* __restrict__ fa,
                         const float* __restrict__ fb,
                         const float* __restrict__ fc,
                         const float* __restrict__ fd,
                         float* __restrict__ out) {
    __shared__ __align__(16) _Float16 sA  [64][LDS_PITCH]; // [b][ re(e) | im(e) ]
    __shared__ __align__(16) _Float16 sBre[64][LDS_PITCH]; // [n][ re(e) | -im(e) ]
    __shared__ __align__(16) _Float16 sBim[64][LDS_PITCH]; // [n][ im(e) |  re(e) ]
    __shared__ float sFC[32][2];                           // core[e]*fa[a][e]

    const int tid     = threadIdx.x;
    const int aIdx    = blockIdx.y;        // 0..63  (row base = aIdx*64)
    const int colBase = blockIdx.x * 64;   // 0..8256

    // ---- Phase 1a: fold core into fa's row for this block ----
    if (tid < 32) {
        float cr = core[2 * tid], ci = core[2 * tid + 1];
        float ar = fa[(aIdx * 32 + tid) * 2], ai = fa[(aIdx * 32 + tid) * 2 + 1];
        sFC[tid][0] = ar * cr - ai * ci;
        sFC[tid][1] = ar * ci + ai * cr;
    }
    __syncthreads();

    // ---- Phase 1b: stage f16 K=64-packed operand tiles in LDS ----
#pragma unroll
    for (int k = 0; k < 8; ++k) {
        int idx = tid + k * 256;     // 0..2047
        int r   = idx >> 5;          // 0..63  (b index / column index)
        int e   = idx & 31;          // rank

        // A side: ab[b=r][e] = (core*fa)[e] * fb[r][e]
        float pr = sFC[e][0], pi = sFC[e][1];
        float br = fb[(r * 32 + e) * 2], bi = fb[(r * 32 + e) * 2 + 1];
        sA[r][e]      = (_Float16)(pr * br - pi * bi);
        sA[r][32 + e] = (_Float16)(pr * bi + pi * br);

        // B side: cd[col][e] = fc[c][e] * fd[d][e], col = c*65+d
        int col = colBase + r;
        int c   = col / 65;
        int d   = col - c * 65;
        float xr = fc[(c * 32 + e) * 2], xi = fc[(c * 32 + e) * 2 + 1];
        float yr = fd[(d * 32 + e) * 2], yi = fd[(d * 32 + e) * 2 + 1];
        float re = xr * yr - xi * yi;
        float im = xr * yi + xi * yr;
        sBre[r][e]      = (_Float16)re;
        sBre[r][32 + e] = (_Float16)(-im);   // out_re = A_re·cd_re + A_im·(-cd_im)
        sBim[r][e]      = (_Float16)im;
        sBim[r][32 + e] = (_Float16)re;      // out_im = A_re·cd_im + A_im·cd_re
    }
    __syncthreads();

    // ---- Phase 2: 8 waves x 2 tiles, 4 WMMA per 16x16 complex tile ----
    const int lane = tid & 31;
    const int half = lane >> 4;   // 0: lanes 0-15, 1: lanes 16-31
    const int lrow = lane & 15;
    const int wave = tid >> 5;

#pragma unroll
    for (int t = 0; t < 2; ++t) {
        int tile = wave * 2 + t;          // 0..15
        int tm = tile >> 2, tn = tile & 3;
        int ar = tm * 16 + lrow;          // A tile row (= M index for this lane)
        int bc = tn * 16 + lrow;          // B tile column (= N index for this lane)

        // A 16x32 f16 frag: lanes 0-15 hold K kb+0..15, lanes 16-31 hold K kb+16..31
        v16h a0 = cat(ld8(&sA[ar][half * 16]),      ld8(&sA[ar][half * 16 + 8]));      // K 0..31
        v16h a1 = cat(ld8(&sA[ar][32 + half * 16]), ld8(&sA[ar][32 + half * 16 + 8]));      // K 32..63

        // B 32x16 f16 frag: lanes 0-15 hold K kb+0..15, lanes 16-31 hold K kb+16..31
        v16h bre0 = cat(ld8(&sBre[bc][half * 16]),      ld8(&sBre[bc][half * 16 + 8]));
        v16h bre1 = cat(ld8(&sBre[bc][32 + half * 16]), ld8(&sBre[bc][32 + half * 16 + 8]));
        v16h bim0 = cat(ld8(&sBim[bc][half * 16]),      ld8(&sBim[bc][half * 16 + 8]));
        v16h bim1 = cat(ld8(&sBim[bc][32 + half * 16]), ld8(&sBim[bc][32 + half * 16 + 8]));

        v8f cre = {};
        v8f cim = {};
        cre = __builtin_amdgcn_wmma_f32_16x16x32_f16(false, a0, false, bre0, (short)0, cre, false, false);
        cre = __builtin_amdgcn_wmma_f32_16x16x32_f16(false, a1, false, bre1, (short)0, cre, false, false);
        cim = __builtin_amdgcn_wmma_f32_16x16x32_f16(false, a0, false, bim0, (short)0, cim, false, false);
        cim = __builtin_amdgcn_wmma_f32_16x16x32_f16(false, a1, false, bim1, (short)0, cim, false, false);

        // C/D layout: VGPR rr, lane<16 -> M=rr ; lane>=16 -> M=rr+8 ; N = lane%16
        long rowG0 = (long)aIdx * 64 + tm * 16 + half * 8;
        int  colG  = colBase + tn * 16 + lrow;
#pragma unroll
        for (int rr = 0; rr < 8; ++rr) {
            long idx = ((rowG0 + rr) * 8320L + colG) * 2;
            v2f v;
            v[0] = cre[rr];
            v[1] = cim[rr];
            __builtin_nontemporal_store(v, reinterpret_cast<v2f*>(out + idx));
        }
    }
}

extern "C" void kernel_launch(void* const* d_in, const int* in_sizes, int n_in,
                              void* d_out, int out_size, void* d_ws, size_t ws_size,
                              hipStream_t stream) {
    (void)in_sizes; (void)n_in; (void)out_size; (void)d_ws; (void)ws_size;
    const float* core = (const float*)d_in[0];  // [32,2]
    const float* fa   = (const float*)d_in[1];  // [64,32,2]
    const float* fb   = (const float*)d_in[2];  // [64,32,2]
    const float* fc   = (const float*)d_in[3];  // [128,32,2]
    const float* fd   = (const float*)d_in[4];  // [65,32,2]
    float* out = (float*)d_out;                 // [64,64,128,65] complex64 interleaved

    dim3 grid(130, 64);   // 8320/64 cols, 4096/64 rows
    cp_reconstruct_wmma<<<grid, 256, 0, stream>>>(core, fa, fb, fc, fd, out);
}
